// GraphAttentionLayer_15307263443517
// MI455X (gfx1250) — matrix operand order repaired and checked by reading, not compiled, not run on hardware
//
#include <hip/hip_runtime.h>
#include <hip/hip_bf16.h>

#define B_DIM 4
#define N_DIM 4096
#define F_DIM 64
#define ALPHA 0.2f
#define NEG_INF -9000000000000000.0f

typedef __attribute__((ext_vector_type(16))) __bf16 v16bf;
typedef __attribute__((ext_vector_type(8)))  __bf16 v8bf;
typedef __attribute__((ext_vector_type(8)))  float  v8f;
typedef __attribute__((ext_vector_type(4)))  float  v4f;
typedef __attribute__((ext_vector_type(4)))  int    v4i;

// ---------------------------------------------------------------------------
// Kernel 1: Wh = h @ W (f32); ei = Wh.a1, ej = Wh.a2; WhT (bf16, [b][f][n]).
// One wave per node row. lane f computes features f and f+32.
// ---------------------------------------------------------------------------
__global__ __launch_bounds__(128) void gat_prep(
    const float* __restrict__ h, const float* __restrict__ W,
    const float* __restrict__ a, float* __restrict__ ei,
    float* __restrict__ ej, unsigned short* __restrict__ whT_raw)
{
    __bf16* whT = (__bf16*)whT_raw;
    const int lane = threadIdx.x & 31;
    const int wv   = threadIdx.x >> 5;
    const int row  = blockIdx.x * 4 + wv;          // global node index b*N+n
    const int b    = row >> 12;                    // row / 4096
    const int n    = row & (N_DIM - 1);

    const float* hrow = h + (size_t)row * F_DIM;
    float acc0 = 0.0f, acc1 = 0.0f;
    #pragma unroll
    for (int k = 0; k < F_DIM; ++k) {
        const float hk = hrow[k];
        acc0 = fmaf(hk, W[k * F_DIM + lane],        acc0);
        acc1 = fmaf(hk, W[k * F_DIM + lane + 32],   acc1);
    }

    // transposed bf16 copy of Wh for WMMA B-fragments
    whT[((size_t)b * F_DIM + lane)      * N_DIM + n] = (__bf16)acc0;
    whT[((size_t)b * F_DIM + lane + 32) * N_DIM + n] = (__bf16)acc1;

    float e1 = acc0 * a[lane]          + acc1 * a[lane + 32];
    float e2 = acc0 * a[F_DIM + lane]  + acc1 * a[F_DIM + lane + 32];
    #pragma unroll
    for (int off = 16; off > 0; off >>= 1) {
        e1 += __shfl_xor(e1, off, 32);
        e2 += __shfl_xor(e2, off, 32);
    }
    if (lane == 0) { ei[row] = e1; ej[row] = e2; }
}

// ---------------------------------------------------------------------------
// Kernel 2: fused masked-softmax attention + att@Wh via bf16 WMMA.
// Block = 4 waves; each wave owns 16 query rows, flash-style loop over j.
// ---------------------------------------------------------------------------
__global__ __launch_bounds__(128) void gat_attn(
    const int* __restrict__ adj, const float* __restrict__ ei,
    const float* __restrict__ ej, const unsigned short* __restrict__ whT_raw,
    float* __restrict__ out)
{
    const __bf16* whT = (const __bf16*)whT_raw;
    const int lane  = threadIdx.x & 31;
    const int wv    = threadIdx.x >> 5;
    const int hi    = lane >> 4;           // half-wave id
    const int lr    = lane & 15;           // row within tile (A-frag), col (B/C-frag)
    const int b     = blockIdx.y;
    const int i0    = (blockIdx.x * 4 + wv) * 16;
    const int row   = i0 + lr;
    const int kbase = hi ? 8 : 0;          // A-fragment K pattern base

    const int*   adj_row = adj + ((size_t)b * N_DIM + row) * (size_t)N_DIM;
    const float* ejb     = ej + b * N_DIM;
    const float  ei_r    = ei[b * N_DIM + row];

    float m = -3.0e38f;                    // running max (per softmax row)
    float l = 0.0f;                        // running denom
    v8f acc[4];
    #pragma unroll
    for (int nt = 0; nt < 4; ++nt)
        #pragma unroll
        for (int r = 0; r < 8; ++r) acc[nt][r] = 0.0f;

    for (int j0 = 0; j0 < N_DIM; j0 += 32) {
        // ---- load adj / ej in the 16-bit A-fragment lane/K pattern:
        // lane<16: K = {0..7, 16..23}; lane>=16: K = {8..15, 24..31}
        const int*   ap = adj_row + j0 + kbase;
        const float* ep = ejb     + j0 + kbase;
        const v4i ad0 = *(const v4i*)(ap);
        const v4i ad1 = *(const v4i*)(ap + 4);
        const v4i ad2 = *(const v4i*)(ap + 16);
        const v4i ad3 = *(const v4i*)(ap + 20);
        const v4f ev0 = *(const v4f*)(ep);
        const v4f ev1 = *(const v4f*)(ep + 4);
        const v4f ev2 = *(const v4f*)(ep + 16);
        const v4f ev3 = *(const v4f*)(ep + 20);

        int   av[16]; float ejv[16];
        #pragma unroll
        for (int t = 0; t < 4; ++t) {
            av[t]      = ad0[t]; av[t + 4]  = ad1[t];
            av[t + 8]  = ad2[t]; av[t + 12] = ad3[t];
            ejv[t]     = ev0[t]; ejv[t + 4]  = ev1[t];
            ejv[t + 8] = ev2[t]; ejv[t + 12] = ev3[t];
        }

        // ---- scores: leaky-relu(ei+ej) masked by adj
        float sv[16];
        #pragma unroll
        for (int t = 0; t < 16; ++t) {
            float e = ei_r + ejv[t];
            e = e > 0.0f ? e : ALPHA * e;
            sv[t] = (av[t] > 0) ? e : NEG_INF;
        }

        // ---- online softmax stats (row split across lane pair L, L+16)
        float lmax = sv[0];
        #pragma unroll
        for (int t = 1; t < 16; ++t) lmax = fmaxf(lmax, sv[t]);
        const float rmax = fmaxf(lmax, __shfl_xor(lmax, 16, 32));
        const float nm   = fmaxf(m, rmax);
        const float sc   = __expf(m - nm);
        m = nm;

        float lsum = 0.0f;
        v16bf af;
        #pragma unroll
        for (int t = 0; t < 16; ++t) {
            const float p = __expf(sv[t] - nm);
            lsum += p;
            af[t] = (__bf16)p;
        }
        lsum += __shfl_xor(lsum, 16, 32);
        l = l * sc + lsum;

        // ---- rescale accumulators: C/D row for vgpr r is (r + 8*hi)
        #pragma unroll
        for (int r = 0; r < 8; ++r) {
            const float scr = __shfl(sc, r + (hi << 3), 32);
            acc[0][r] *= scr; acc[1][r] *= scr;
            acc[2][r] *= scr; acc[3][r] *= scr;
        }

        // ---- B fragments (32x16 bf16: lane = column, K = 8*hi + {0..7, 16..23})
        #pragma unroll
        for (int nt = 0; nt < 4; ++nt) {
            const __bf16* bp =
                whT + ((size_t)b * F_DIM + lr + nt * 16) * (size_t)N_DIM
                    + j0 + hi * 8;
            const v8bf blo = *(const v8bf*)(bp);
            const v8bf bhi = *(const v8bf*)(bp + 16);
            v16bf bf;
            #pragma unroll
            for (int t = 0; t < 8; ++t) { bf[t] = blo[t]; bf[t + 8] = bhi[t]; }
            acc[nt] = __builtin_amdgcn_wmma_f32_16x16x32_bf16(
                false, af, false, bf, (short)0, acc[nt], false, false);
        }
    }

    // ---- normalize + ELU + store
    const float inv = 1.0f / l;
    #pragma unroll
    for (int r = 0; r < 8; ++r) {
        const float invr = __shfl(inv, r + (hi << 3), 32);
        const int   orow = i0 + r + hi * 8;
        float* op = out + ((size_t)b * N_DIM + orow) * F_DIM + lr;
        #pragma unroll
        for (int nt = 0; nt < 4; ++nt) {
            float v = acc[nt][r] * invr;
            v = v > 0.0f ? v : (__expf(v) - 1.0f);
            op[nt * 16] = v;
        }
    }
}

// ---------------------------------------------------------------------------
extern "C" void kernel_launch(void* const* d_in, const int* in_sizes, int n_in,
                              void* d_out, int out_size, void* d_ws, size_t ws_size,
                              hipStream_t stream) {
    const float* h   = (const float*)d_in[0];
    const int*   adj = (const int*)d_in[1];
    const float* W   = (const float*)d_in[2];
    const float* a   = (const float*)d_in[3];
    float*       out = (float*)d_out;

    char* ws = (char*)d_ws;
    float* ei = (float*)ws;                                   // B*N floats
    float* ej = ei + B_DIM * N_DIM;                           // B*N floats
    unsigned short* whT = (unsigned short*)(ws + 2ull * B_DIM * N_DIM * sizeof(float));

    gat_prep<<<B_DIM * N_DIM / 4, 128, 0, stream>>>(h, W, a, ei, ej, whT);
    gat_attn<<<dim3(N_DIM / 64, B_DIM), 128, 0, stream>>>(adj, ei, ej, whT, out);
}
